// MLA_8246337208875
// MI455X (gfx1250) — matrix operand order repaired and checked by reading, not compiled, not run on hardware
//
#include <hip/hip_runtime.h>
#include <hip/hip_bf16.h>
#include <math.h>

// ---------------------------------------------------------------------------
// MLA layer for MI455X (gfx1250).
// All contractions via v_wmma_f32_16x16x32_bf16; all GEMM/attention tile
// staging via CDNA5 async global->LDS DMA (global_load_async_to_lds_b128,
// ASYNCcnt). Operands pre-converted to bf16 once (halves HBM traffic).
// GEMM: 256x128 workgroup tile, 8 waves of 64x64 -> 16 WMMA : 16 ds_load_b128.
// ---------------------------------------------------------------------------

typedef __attribute__((ext_vector_type(16))) __bf16 v16bf;
typedef __attribute__((ext_vector_type(8)))  float  v8f;

union FragU { unsigned int u[8]; v16bf v; };

__device__ __forceinline__ unsigned short f2bf(float f) {
  unsigned int u = __float_as_uint(f);
  unsigned int r = u + 0x7FFFu + ((u >> 16) & 1u);   // round-to-nearest-even
  return (unsigned short)(r >> 16);
}

__device__ __forceinline__ v8f wmma_bf16(v16bf a, v16bf b, v8f c) {
  return __builtin_amdgcn_wmma_f32_16x16x32_bf16(false, a, false, b, (short)0, c, false, false);
}

// CDNA5 async tensor/LDS path: 16B global -> LDS, tracked by ASYNCcnt.
__device__ __forceinline__ void async_b128(unsigned lds_off, unsigned long long gbase,
                                           unsigned goff) {
  asm volatile("global_load_async_to_lds_b128 %0, %1, %2"
               :: "v"(lds_off), "v"(goff), "s"(gbase) : "memory");
}
__device__ __forceinline__ void wait_async0() {
  asm volatile("s_wait_asynccnt 0x0" ::: "memory");
}

// A-fragment (16x32, lane = row): v0..3 = K[8*half..+7], v4..7 = K[16+8*half..+7]
__device__ __forceinline__ v16bf fragA_lds(const unsigned short* lds, int row, int pitch,
                                           int kbase, int half) {
  const unsigned int* p = (const unsigned int*)(lds + row * pitch + kbase);
  FragU f;
#pragma unroll
  for (int j = 0; j < 4; ++j) f.u[j]     = p[half * 4 + j];
#pragma unroll
  for (int j = 0; j < 4; ++j) f.u[4 + j] = p[8 + half * 4 + j];
  return f.v;
}

// B-fragment (32x16, lane = col): v0..3 = K[8*half..+7], v4..7 = K[16+8*half..+7]
__device__ __forceinline__ v16bf fragB_lds(const unsigned short* lds, int col, int pitch,
                                           int kbase, int half) {
  const unsigned int* p = (const unsigned int*)(lds + col * pitch + kbase);
  FragU f;
#pragma unroll
  for (int j = 0; j < 4; ++j) f.u[j]     = p[half * 4 + j];
#pragma unroll
  for (int j = 0; j < 4; ++j) f.u[4 + j] = p[8 + half * 4 + j];
  return f.v;
}

// A-fragment straight from global bf16 (row-major, K contiguous)
__device__ __forceinline__ v16bf fragA_g16(const unsigned short* rowp, int kbase, int half) {
  FragU f;
  const uint4 a = *(const uint4*)(rowp + kbase + 8 * half);
  const uint4 b = *(const uint4*)(rowp + kbase + 16 + 8 * half);
  f.u[0] = a.x; f.u[1] = a.y; f.u[2] = a.z; f.u[3] = a.w;
  f.u[4] = b.x; f.u[5] = b.y; f.u[6] = b.z; f.u[7] = b.w;
  return f.v;
}

// ---------------------------------------------------------------------------
// Elementwise f32 -> bf16 (n % 1024 == 0)
// ---------------------------------------------------------------------------
__global__ __launch_bounds__(256) void convert_f32_bf16_kernel(
    const float* __restrict__ in, unsigned short* __restrict__ out) {
  const size_t i = ((size_t)blockIdx.x * 256 + threadIdx.x) * 4;
  const float4 f = *(const float4*)(in + i);
  uint2 p;
  p.x = (unsigned)f2bf(f.x) | ((unsigned)f2bf(f.y) << 16);
  p.y = (unsigned)f2bf(f.z) | ((unsigned)f2bf(f.w) << 16);
  *(uint2*)(out + i) = p;
}

// ---------------------------------------------------------------------------
// Transpose+convert: in[K,N] f32 -> out[N,K] bf16 (K,N multiples of 32)
// ---------------------------------------------------------------------------
__global__ __launch_bounds__(256) void transpose_f32_bf16_kernel(
    const float* __restrict__ in, unsigned short* __restrict__ out, int K, int N) {
  __shared__ float t[32][33];
  const int n0 = blockIdx.x * 32, k0 = blockIdx.y * 32;
  const int tx = threadIdx.x & 31, ty = threadIdx.x >> 5;
#pragma unroll
  for (int i = 0; i < 32; i += 8)
    t[ty + i][tx] = in[(size_t)(k0 + ty + i) * N + n0 + tx];
  __syncthreads();
#pragma unroll
  for (int i = 0; i < 32; i += 8)
    out[(size_t)(n0 + ty + i) * K + k0 + tx] = f2bf(t[tx][ty + i]);
}

// ---------------------------------------------------------------------------
// GEMM: C[M,N](f32) = A[M,K](bf16) * Bt[N,K](bf16, pre-transposed)
// 256x128 block, 8 waves (4x2), wave tile 64x64 (4x4 accum tiles).
// Double-buffered LDS filled by async DMA; copy of tile k+1 overlaps WMMA of
// tile k. Requires M%256==0, K%32==0.
// ---------------------------------------------------------------------------
#define GP 40   // LDS pitch in bf16 elems (32 + 8 pad; 80B rows, 16B aligned)

__global__ __launch_bounds__(256) void gemm_bf16_kernel(
    const unsigned short* __restrict__ A, const unsigned short* __restrict__ Bt,
    float* __restrict__ C, int M, int N, int K) {
  __shared__ __align__(16) unsigned short Alds[2][256 * GP];
  __shared__ __align__(16) unsigned short Blds[2][128 * GP];

  const int tid = threadIdx.x, wid = tid >> 5, lane = tid & 31;
  const int half = lane >> 4, lr = lane & 15;
  const int m0 = blockIdx.y * 256, n0 = blockIdx.x * 128;
  const int wm = wid >> 1, wn = wid & 1;        // 4 x 2 waves

  const unsigned long long aG = (unsigned long long)A;
  const unsigned long long bG = (unsigned long long)Bt;
  const unsigned aL[2] = { (unsigned)(size_t)&Alds[0][0], (unsigned)(size_t)&Alds[1][0] };
  const unsigned bL[2] = { (unsigned)(size_t)&Blds[0][0], (unsigned)(size_t)&Blds[1][0] };

  // copy jobs: A -> each thread owns row tid (4 chunks of 16B)
  //            B -> each thread owns row tid>>1, two 16B chunks
  const int br  = tid >> 1;
  const int bc0 = (tid & 1) * 2;
  const bool brow_ok = (n0 + br) < N;

  auto issue_tile = [&](int kt, int buf) {
    const int k0 = kt << 5;
#pragma unroll
    for (int ch = 0; ch < 4; ++ch)
      async_b128(aL[buf] + (unsigned)(tid * GP * 2 + ch * 16), aG,
                 (unsigned)(((size_t)(m0 + tid) * K + k0 + ch * 8) * 2));
    if (brow_ok) {
#pragma unroll
      for (int cc = 0; cc < 2; ++cc) {
        const int ch = bc0 + cc;
        async_b128(bL[buf] + (unsigned)(br * GP * 2 + ch * 16), bG,
                   (unsigned)(((size_t)(n0 + br) * K + k0 + ch * 8) * 2));
      }
    }
  };

  v8f acc[4][4];
#pragma unroll
  for (int i = 0; i < 4; ++i)
#pragma unroll
    for (int j = 0; j < 4; ++j)
      acc[i][j] = v8f{0.f, 0.f, 0.f, 0.f, 0.f, 0.f, 0.f, 0.f};

  const int nk = K >> 5;
  issue_tile(0, 0);
  for (int kt = 0; kt < nk; ++kt) {
    wait_async0();
    __syncthreads();                       // tile kt in LDS; buf^1 free again
    if (kt + 1 < nk) issue_tile(kt + 1, (kt + 1) & 1);
    const unsigned short* Ab = Alds[kt & 1];
    const unsigned short* Bb = Blds[kt & 1];
    v16bf af[4], bfr[4];
#pragma unroll
    for (int i = 0; i < 4; ++i) af[i]  = fragA_lds(Ab, wm * 64 + i * 16 + lr, GP, 0, half);
#pragma unroll
    for (int j = 0; j < 4; ++j) bfr[j] = fragB_lds(Bb, wn * 64 + j * 16 + lr, GP, 0, half);
#pragma unroll
    for (int i = 0; i < 4; ++i)
#pragma unroll
      for (int j = 0; j < 4; ++j)
        acc[i][j] = wmma_bf16(af[i], bfr[j], acc[i][j]);
  }

#pragma unroll
  for (int i = 0; i < 4; ++i) {
    const int rowb = m0 + wm * 64 + i * 16 + half * 8;
#pragma unroll
    for (int j = 0; j < 4; ++j) {
      const int col = n0 + wn * 64 + j * 16 + lr;
      if (col < N) {
#pragma unroll
        for (int e = 0; e < 8; ++e)
          C[(size_t)(rowb + e) * N + col] = acc[i][j][e];
      }
    }
  }
}

// ---------------------------------------------------------------------------
// RMSNorm: f32 in -> bf16 out, one block per row
// ---------------------------------------------------------------------------
__global__ __launch_bounds__(256) void rmsnorm_bf16_kernel(
    const float* __restrict__ in, const float* __restrict__ w,
    unsigned short* __restrict__ out, int width, int istride, int ostride) {
  __shared__ float red[256];
  const int row = blockIdx.x, tid = threadIdx.x;
  const float* ip = in + (size_t)row * istride;
  float s = 0.f;
  for (int i = tid; i < width; i += 256) { float v = ip[i]; s += v * v; }
  red[tid] = s;
  __syncthreads();
  for (int off = 128; off > 0; off >>= 1) {
    if (tid < off) red[tid] += red[tid + off];
    __syncthreads();
  }
  const float scale = rsqrtf(red[0] / (float)width + 1e-6f);
  unsigned short* op = out + (size_t)row * ostride;
  for (int i = tid; i < width; i += 256) op[i] = f2bf(ip[i] * scale * w[i]);
}

// ---------------------------------------------------------------------------
// YaRN RoPE + pack to bf16: q16[B*L,16,192], k16[B*L,16,192], v16[B,16,128,L]
// ---------------------------------------------------------------------------
__global__ __launch_bounds__(256) void rope_pack_kernel(
    const float* __restrict__ Qf,    // [B*L,16,192]
    const float* __restrict__ KVA,   // [B*L,576]
    const float* __restrict__ KVB,   // [B*L,16,256]
    unsigned short* __restrict__ Q16,
    unsigned short* __restrict__ K16,
    unsigned short* __restrict__ V16,
    int L) {
  __shared__ float cs[32], sn[32], kr[64];
  const int row = blockIdx.x;
  const int pos = row % L;
  const int b   = row / L;
  const int tid = threadIdx.x;

  if (tid < 32) {
    float p2 = 2.0f * (float)tid;
    float inv_freq = powf(10000.0f, -(p2 / 64.0f));
    float wavelen = 6.28318530717958647f / inv_freq;
    float ramp = (8192.0f / wavelen - 1.0f) / 3.0f;
    ramp = fminf(fmaxf(ramp, 0.0f), 1.0f);
    inv_freq = inv_freq * (1.0f - ramp) + (inv_freq / 16.0f) * ramp;
    float fr = (float)pos * inv_freq;
    float af = 0.1f * logf(16.0f) + 1.0f;
    cs[tid] = cosf(fr) * af;
    sn[tid] = sinf(fr) * af;
  }
  __syncthreads();
  if (tid < 32) {
    float x0 = KVA[(size_t)row * 576 + 512 + tid];
    float x1 = KVA[(size_t)row * 576 + 512 + 32 + tid];
    kr[tid]      = x0 * cs[tid] - x1 * sn[tid];
    kr[tid + 32] = x1 * cs[tid] + x0 * sn[tid];
  }
  __syncthreads();

  // q_nope passthrough
  for (int idx = tid; idx < 16 * 128; idx += 256) {
    int hh = idx >> 7, d = idx & 127;
    size_t o = ((size_t)row * 16 + hh) * 192 + d;
    Q16[o] = f2bf(Qf[o]);
  }
  // q_rope
  for (int idx = tid; idx < 16 * 32; idx += 256) {
    int hh = idx >> 5, i = idx & 31;
    size_t base = ((size_t)row * 16 + hh) * 192 + 128;
    float x0 = Qf[base + i], x1 = Qf[base + i + 32];
    Q16[base + i]      = f2bf(x0 * cs[i] - x1 * sn[i]);
    Q16[base + i + 32] = f2bf(x1 * cs[i] + x0 * sn[i]);
  }
  // k = [k_nope | roped k_rope] broadcast over heads
  for (int idx = tid; idx < 16 * 192; idx += 256) {
    int hh = idx / 192, d = idx % 192;
    float v = (d < 128) ? KVB[((size_t)row * 16 + hh) * 256 + d] : kr[d - 128];
    K16[((size_t)row * 16 + hh) * 192 + d] = f2bf(v);
  }
  // v, transposed to [B,16,128,L] so 32 keys are one contiguous 64B chunk
  for (int idx = tid; idx < 16 * 128; idx += 256) {
    int hh = idx >> 7, d = idx & 127;
    V16[(((size_t)b * 16 + hh) * 128 + d) * L + pos] =
        f2bf(KVB[((size_t)row * 16 + hh) * 256 + 128 + d]);
  }
}

// ---------------------------------------------------------------------------
// Causal flash attention (bf16 in/out, f32 softmax+accum in VGPRs).
// Block = 128 query rows of one (b,h); 8 waves x 16 rows. K/V tiles via async DMA.
// ---------------------------------------------------------------------------
#define KP 200   // Klds pitch (192+8 shorts; 400B rows, 16B aligned)
#define VP 40    // Vlds/Plds pitch (32+8 shorts; 80B rows, 16B aligned)

__global__ __launch_bounds__(256) void mla_attn_kernel(
    const unsigned short* __restrict__ Q,   // [B*L,16,192] bf16
    const unsigned short* __restrict__ Kb,  // [B*L,16,192] bf16
    const unsigned short* __restrict__ Vt,  // [B,16,128,L] bf16
    unsigned short* __restrict__ O,         // [B*L,16,128] bf16 out
    int L) {
  __shared__ __align__(16) unsigned short Klds[32 * KP];
  __shared__ __align__(16) unsigned short Vlds[128 * VP];
  __shared__ __align__(16) unsigned short Plds[8 * 16 * VP];

  const int tid = threadIdx.x, wid = tid >> 5, lane = tid & 31;
  const int half = lane >> 4, lr = lane & 15;
  const int q0 = blockIdx.x * 128;
  const int b = blockIdx.y >> 4, h = blockIdx.y & 15;

  const unsigned long long kG = (unsigned long long)Kb;
  const unsigned long long vG = (unsigned long long)Vt;
  const unsigned kLB = (unsigned)(size_t)&Klds[0];
  const unsigned vLB = (unsigned)(size_t)&Vlds[0];

  // Q fragments: 6 chunks of 16x32, straight from global bf16
  const int qrow = q0 + wid * 16 + lr;
  const unsigned short* qp = Q + ((size_t)(b * L + qrow) * 16 + h) * 192;
  v16bf qf[6];
#pragma unroll
  for (int c = 0; c < 6; ++c) qf[c] = fragA_g16(qp, c * 32, half);

  float rowm[8], rowl[8];
  v8f oacc[8];
#pragma unroll
  for (int j = 0; j < 8; ++j) { rowm[j] = -1e30f; rowl[j] = 0.f; }
#pragma unroll
  for (int vt = 0; vt < 8; ++vt) oacc[vt] = v8f{0.f, 0.f, 0.f, 0.f, 0.f, 0.f, 0.f, 0.f};

  const float scale = rsqrtf(192.0f);
  const int nkb = (q0 + 128) >> 5;            // causal

  for (int kb = 0; kb < nkb; ++kb) {
    const int k0 = kb << 5;
    __syncthreads();                           // previous compute done reading LDS
    // K block: 32 keys x 384B -> 768 16B chunks
    for (int idx = tid; idx < 768; idx += 256) {
      int key = idx / 24, c = idx % 24;
      async_b128(kLB + (unsigned)(key * KP * 2 + c * 16), kG,
                 (unsigned)((((size_t)(b * L + k0 + key) * 16 + h) * 192) * 2 + c * 16));
    }
    // V block: 128 d-rows x 64B -> 512 16B chunks
    for (int idx = tid; idx < 512; idx += 256) {
      int d = idx >> 2, c = idx & 3;
      async_b128(vLB + (unsigned)(d * VP * 2 + c * 16), vG,
                 (unsigned)(((((size_t)b * 16 + h) * 128 + d) * L + k0) * 2 + c * 16));
    }
    wait_async0();
    __syncthreads();

    // scores: two 16x16 tiles over 32 keys
    v8f s0 = v8f{0.f, 0.f, 0.f, 0.f, 0.f, 0.f, 0.f, 0.f};
    v8f s1 = v8f{0.f, 0.f, 0.f, 0.f, 0.f, 0.f, 0.f, 0.f};
#pragma unroll
    for (int c = 0; c < 6; ++c) {
      v16bf b0 = fragB_lds(Klds, lr,      KP, c * 32, half);
      v16bf b1 = fragB_lds(Klds, 16 + lr, KP, c * 32, half);
      s0 = wmma_bf16(qf[c], b0, s0);
      s1 = wmma_bf16(qf[c], b1, s1);
    }

    unsigned short* pl = Plds + wid * 16 * VP;
#pragma unroll
    for (int j = 0; j < 8; ++j) {
      const int qrj = q0 + wid * 16 + half * 8 + j;
      float a0 = s0[j] * scale, a1 = s1[j] * scale;
      if (k0 + lr > qrj)      a0 = -1e30f;
      if (k0 + 16 + lr > qrj) a1 = -1e30f;
      float t = fmaxf(a0, a1);
      t = fmaxf(t, __shfl_xor(t, 1, 16));
      t = fmaxf(t, __shfl_xor(t, 2, 16));
      t = fmaxf(t, __shfl_xor(t, 4, 16));
      t = fmaxf(t, __shfl_xor(t, 8, 16));
      const float mnew = fmaxf(rowm[j], t);
      const float corr = __expf(rowm[j] - mnew);
      rowm[j] = mnew;
      const float p0 = __expf(a0 - mnew), p1 = __expf(a1 - mnew);
      float ps = p0 + p1;
      ps += __shfl_xor(ps, 1, 16);
      ps += __shfl_xor(ps, 2, 16);
      ps += __shfl_xor(ps, 4, 16);
      ps += __shfl_xor(ps, 8, 16);
      rowl[j] = rowl[j] * corr + ps;
#pragma unroll
      for (int vt = 0; vt < 8; ++vt) oacc[vt][j] *= corr;
      pl[(half * 8 + j) * VP + lr]      = f2bf(p0);
      pl[(half * 8 + j) * VP + 16 + lr] = f2bf(p1);
    }
    asm volatile("s_wait_dscnt 0x0" ::: "memory");   // wave-local P RAW

    v16bf pa = fragA_lds(pl, lr, VP, 0, half);
#pragma unroll
    for (int vt = 0; vt < 8; ++vt) {
      v16bf vb = fragB_lds(Vlds, vt * 16 + lr, VP, 0, half);
      oacc[vt] = wmma_bf16(pa, vb, oacc[vt]);
    }
  }

#pragma unroll
  for (int j = 0; j < 8; ++j) {
    const float inv = 1.0f / rowl[j];
    const int qrj = q0 + wid * 16 + half * 8 + j;
    unsigned short* op = O + ((size_t)(b * L + qrj) * 16 + h) * 128;
#pragma unroll
    for (int vt = 0; vt < 8; ++vt) op[vt * 16 + lr] = f2bf(oacc[vt][j] * inv);
  }
}

// ---------------------------------------------------------------------------
// Host-side orchestration
// ---------------------------------------------------------------------------
extern "C" void kernel_launch(void* const* d_in, const int* in_sizes, int n_in,
                              void* d_out, int out_size, void* d_ws, size_t ws_size,
                              hipStream_t stream) {
  const float* x         = (const float*)d_in[0];  // [4096,2048]
  const float* w_qa      = (const float*)d_in[1];  // [2048,1536]
  const float* q_norm_w  = (const float*)d_in[2];  // [1536]
  const float* w_qb      = (const float*)d_in[3];  // [1536,3072]
  const float* w_kva     = (const float*)d_in[4];  // [2048,576]
  const float* kv_norm_w = (const float*)d_in[5];  // [512]
  const float* w_kvb     = (const float*)d_in[6];  // [512,4096]
  const float* w_o       = (const float*)d_in[7];  // [2048,2048]
  float* out = (float*)d_out;                      // [4096,2048]
  (void)in_sizes; (void)n_in; (void)out_size; (void)ws_size;

  const int Bb = 2, L = 2048, M = Bb * L;          // M = 4096

  char* ws = (char*)d_ws;
  size_t off = 0;
  auto alloc_f = [&](size_t n) { float* p = (float*)(ws + off);
    off = (off + n * 4 + 255) & ~(size_t)255; return p; };
  auto alloc_h = [&](size_t n) { unsigned short* p = (unsigned short*)(ws + off);
    off = (off + n * 2 + 255) & ~(size_t)255; return p; };

  unsigned short* xb16    = alloc_h((size_t)M * 2048);
  unsigned short* wt_qa   = alloc_h((size_t)1536 * 2048);
  unsigned short* wt_qb   = alloc_h((size_t)3072 * 1536);
  unsigned short* wt_kva  = alloc_h((size_t)576 * 2048);
  unsigned short* wt_kvb  = alloc_h((size_t)4096 * 512);
  unsigned short* wt_o    = alloc_h((size_t)2048 * 2048);
  float*          q_lora  = alloc_f((size_t)M * 1536);
  unsigned short* q_l16   = alloc_h((size_t)M * 1536);
  float*          qbuf    = alloc_f((size_t)M * 3072);
  float*          kva     = alloc_f((size_t)M * 576);
  unsigned short* kvn16   = alloc_h((size_t)M * 512);
  float*          kvb     = alloc_f((size_t)M * 4096);
  unsigned short* q16     = alloc_h((size_t)M * 3072);
  unsigned short* k16     = alloc_h((size_t)M * 3072);
  unsigned short* v16     = alloc_h((size_t)Bb * 16 * 128 * L);
  unsigned short* attn16  = alloc_h((size_t)M * 2048);

  const dim3 blk(256);

  // prologue: convert x; transpose+convert weights (one pass each)
  convert_f32_bf16_kernel<<<(M * 2048) / 1024, blk, 0, stream>>>(x, xb16);
  transpose_f32_bf16_kernel<<<dim3(1536 / 32, 2048 / 32), blk, 0, stream>>>(w_qa,  wt_qa,  2048, 1536);
  transpose_f32_bf16_kernel<<<dim3(3072 / 32, 1536 / 32), blk, 0, stream>>>(w_qb,  wt_qb,  1536, 3072);
  transpose_f32_bf16_kernel<<<dim3(576  / 32, 2048 / 32), blk, 0, stream>>>(w_kva, wt_kva, 2048, 576);
  transpose_f32_bf16_kernel<<<dim3(4096 / 32, 512  / 32), blk, 0, stream>>>(w_kvb, wt_kvb, 512,  4096);
  transpose_f32_bf16_kernel<<<dim3(2048 / 32, 2048 / 32), blk, 0, stream>>>(w_o,   wt_o,   2048, 2048);

  // q path
  gemm_bf16_kernel<<<dim3(1536 / 128, M / 256), blk, 0, stream>>>(xb16, wt_qa, q_lora, M, 1536, 2048);
  rmsnorm_bf16_kernel<<<M, blk, 0, stream>>>(q_lora, q_norm_w, q_l16, 1536, 1536, 1536);
  gemm_bf16_kernel<<<dim3(3072 / 128, M / 256), blk, 0, stream>>>(q_l16, wt_qb, qbuf, M, 3072, 1536);
  // kv path
  gemm_bf16_kernel<<<dim3((576 + 127) / 128, M / 256), blk, 0, stream>>>(xb16, wt_kva, kva, M, 576, 2048);
  rmsnorm_bf16_kernel<<<M, blk, 0, stream>>>(kva, kv_norm_w, kvn16, 512, 576, 512);
  gemm_bf16_kernel<<<dim3(4096 / 128, M / 256), blk, 0, stream>>>(kvn16, wt_kvb, kvb, M, 4096, 512);
  // rope + pack to bf16
  rope_pack_kernel<<<M, blk, 0, stream>>>(qbuf, kva, kvb, q16, k16, v16, L);
  // attention
  mla_attn_kernel<<<dim3(L / 128, Bb * 16), blk, 0, stream>>>(q16, k16, v16, attn16, L);
  // output projection
  gemm_bf16_kernel<<<dim3(2048 / 128, M / 256), blk, 0, stream>>>(attn16, wt_o, out, M, 2048, 2048);
}
